// MultidimensionalLSTM_867583394651
// MI455X (gfx1250) — matrix operand order repaired and checked by reading, not compiled, not run on hardware
//
#include <hip/hip_runtime.h>
#include <hip/hip_bf16.h>

// ---------------------------------------------------------------------------
// LSTM (B=128, T=512, F=64, H=512) + Dense(64) for MI455X (gfx1250, wave32).
// bf16 WMMA (v_wmma_f32_16x16x32_bf16) + TDM (tensor_load_to_lds) prefetch.
// ---------------------------------------------------------------------------

typedef __attribute__((ext_vector_type(16))) __bf16       v16bf;
typedef __attribute__((ext_vector_type(8)))  float        v8f;
typedef __attribute__((ext_vector_type(4)))  unsigned int vu4;
typedef __attribute__((ext_vector_type(4)))  int          v4i;
typedef __attribute__((ext_vector_type(8)))  int          v8i;

#define Bsz  128
#define Tsz  512
#define Fsz  64
#define Hsz  512
#define G4H  2048
#define FUT  64
#define HS   536     // padded LDS h row stride (elems): 16B-aligned, conflict-free

// LDS layout (dynamic, base assumed 0: no static __shared__ in this TU)
#define XP_ELEMS   (16 * G4H)                 // 32768 bf16 per buffer
#define XP_BYTES   (XP_ELEMS * 2)             // 64 KB
#define LDS_XP0    0
#define LDS_XP1    XP_BYTES
#define LDS_H      (2 * XP_BYTES)             // h double buffer after xp buffers
#define LDS_TOTAL  (2 * XP_BYTES + 2 * 16 * HS * 2)

#if __has_builtin(__builtin_amdgcn_tensor_load_to_lds)
#define HAVE_TDM 1
#else
#define HAVE_TDM 0
#endif

struct bfpack { vu4 lo, hi; };

// Build a v16bf WMMA operand from two 16-byte chunks.
__device__ __forceinline__ v16bf ld2(const __bf16* p0, const __bf16* p1) {
  bfpack t;
  t.lo = *(const vu4*)p0;
  t.hi = *(const vu4*)p1;
  return __builtin_bit_cast(v16bf, t);
}

__device__ __forceinline__ float sigmoidf(float x) {
  return 1.0f / (1.0f + __expf(-x));
}

#if HAVE_TDM
// 1-D TDM copy: 32768 bf16 elements, contiguous, global -> LDS.
__device__ __forceinline__ void tdm_load_xp(const __bf16* gsrc, unsigned lds_off) {
  unsigned long long ga = (unsigned long long)gsrc;
  vu4 g0 = { 1u,                                   // count=1 (valid user D#)
             lds_off,                              // lds_addr (bytes)
             (unsigned)(ga & 0xFFFFFFFFu),         // global_addr[31:0]
             (unsigned)((ga >> 32) & 0x1FFFFFFu) | (2u << 30) }; // addr[56:32] | type=2
  v8i g1 = { (int)(1u << 16),                      // wg_mask=0, data_size=1 (2B)
             (int)0x80000000u,                     // tensor_dim0[15:0]=32768 (<<16)
             (int)0x00010000u,                     // tensor_dim1=1
             (int)0x80000000u,                     // tile_dim0=32768 (<<16)
             0,                                    // tile_dim1=0, tile_dim2=0
             32768,                                // tensor_dim0_stride
             0, 0 };
  v4i z4 = { 0, 0, 0, 0 };
#if __clang_major__ >= 23
  v8i z8 = { 0, 0, 0, 0, 0, 0, 0, 0 };
  __builtin_amdgcn_tensor_load_to_lds(g0, g1, z4, z4, z8, 0);
#else
  __builtin_amdgcn_tensor_load_to_lds(g0, g1, z4, z4, 0);
#endif
}
#endif

// ---------------------------------------------------------------------------
// Conversion kernels: fp32 -> bf16 (plain and transposed).
// ---------------------------------------------------------------------------
__global__ void k_cvt(const float* __restrict__ src, __bf16* __restrict__ dst, int n) {
  int i = blockIdx.x * 256 + threadIdx.x;
  if (i < n) dst[i] = (__bf16)src[i];
}

// src[rows][cols] fp32 -> dst[cols][rows] bf16
__global__ void k_cvt_t(const float* __restrict__ src, __bf16* __restrict__ dst,
                        int rows, int cols) {
  int i = blockIdx.x * 256 + threadIdx.x;
  if (i < rows * cols) {
    int r = i / cols, c = i % cols;
    dst[(long)c * rows + r] = (__bf16)src[i];
  }
}

// ---------------------------------------------------------------------------
// x_proj[t*B+b][n] = x[b][t][:] @ W[:][n] + bias[n], stored bf16, [T][B][4H].
// One 16x16 output tile per wave; K=64 -> 2 WMMAs, bias-seeded accumulator.
// ---------------------------------------------------------------------------
__global__ __launch_bounds__(256)
void k_xproj(const __bf16* __restrict__ xb,   // [B][T][F] bf16
             const __bf16* __restrict__ Wt,   // [4H][F]  bf16 (transposed W)
             const float*  __restrict__ bias, // [4H]
             __bf16*       __restrict__ xproj) {
  const int wave = threadIdx.x >> 5;
  const int lane = threadIdx.x & 31;
  const int lo = lane & 15, hi = lane >> 4;

  const int NT = G4H / 16;                        // 128 column tiles
  long tile = (long)blockIdx.x * 8 + wave;        // 524288 tiles total
  int nt = (int)(tile % NT);
  int mt = (int)(tile / NT);
  int row0 = mt * 16;                             // global row = t*B + b
  int t  = row0 / Bsz;
  int b0 = row0 % Bsz;                            // 16-row block never crosses B
  int n0 = nt * 16;

  const __bf16* arow = xb + ((long)(b0 + lo) * Tsz + t) * Fsz;
  const __bf16* bcol = Wt + (long)(n0 + lo) * Fsz;

  v8f acc;
  float bv = bias[n0 + lo];
#pragma unroll
  for (int r = 0; r < 8; ++r) acc[r] = bv;

#pragma unroll
  for (int kb = 0; kb < 2; ++kb) {
    v16bf a = ld2(arow + kb * 32 + hi * 16, arow + kb * 32 + hi * 16 + 8);
    v16bf w = ld2(bcol + kb * 32 + hi * 16, bcol + kb * 32 + hi * 16 + 8);
    acc = __builtin_amdgcn_wmma_f32_16x16x32_bf16(false, a, false, w,
                                                  (short)0, acc, false, false);
  }

  __bf16* orow = xproj + (long)row0 * G4H + n0;
#pragma unroll
  for (int r = 0; r < 8; ++r)
    orow[(long)(r + 8 * hi) * G4H + lo] = (__bf16)acc[r];
}

// ---------------------------------------------------------------------------
// Persistent recurrent scan. 8 blocks (one per 16 batch rows), 1024 threads
// = 32 waves; wave w owns h columns [16w,16w+16) across all four gates, so
// c and the gate math stay in registers. h double-buffered in LDS; the next
// step's x_proj tile (contiguous 64 KB) is TDM-prefetched into LDS by wave 0
// while everyone runs the WMMA chain.
// ---------------------------------------------------------------------------
__global__ __launch_bounds__(1024)
void k_scan(const __bf16* __restrict__ xproj, // [T][B][4H] bf16
            const __bf16* __restrict__ Ut,    // [4H][H] bf16 (transposed U)
            __bf16*       __restrict__ hbf)   // [B][H] bf16 (final h)
{
  extern __shared__ char smem[];
  __bf16* xp   = (__bf16*)(smem + LDS_XP0);   // 2 x 32768 bf16
  __bf16* hbuf = (__bf16*)(smem + LDS_H);     // 2 x 16 x HS bf16

  const int wave = threadIdx.x >> 5;
  const int lane = threadIdx.x & 31;
  const int lo = lane & 15, hi = lane >> 4;
  const int b0 = blockIdx.x * 16;             // batch rows owned by this block
  const int n0 = wave * 16;                   // h columns owned by this wave

  // Prologue: fetch x_proj tile for t=0 and zero h(0).
#if HAVE_TDM
  if (wave == 0)
    tdm_load_xp(xproj + (long)b0 * G4H, LDS_XP0);
#else
  {
    const __bf16* src = xproj + (long)b0 * G4H;
    for (int i = threadIdx.x * 8; i < XP_ELEMS; i += 1024 * 8)
      *(vu4*)(xp + i) = *(const vu4*)(src + i);
  }
#endif
  for (int i = threadIdx.x; i < 16 * HS; i += 1024) hbuf[i] = (__bf16)0.0f;
#if HAVE_TDM
  if (wave == 0) __builtin_amdgcn_s_wait_tensorcnt(0);
#endif
  __syncthreads();

  v8f c;
#pragma unroll
  for (int r = 0; r < 8; ++r) c[r] = 0.0f;

  // Per-lane B-operand base pointers for the 4 gate column slices of U^T.
  const __bf16* up[4];
#pragma unroll
  for (int g = 0; g < 4; ++g)
    up[g] = Ut + (long)(g * Hsz + n0 + lo) * Hsz + hi * 16;

#pragma unroll 1
  for (int t = 0; t < Tsz; ++t) {
    const int pcur = t & 1, pnxt = (t + 1) & 1;
    const __bf16* cur   = hbuf + pcur * (16 * HS);
    __bf16*       nxt   = hbuf + pnxt * (16 * HS);
    const __bf16* xpcur = xp + pcur * XP_ELEMS;

    // Seed accumulators with x_proj (input GEMM + bias already applied).
    v8f acc[4];
#pragma unroll
    for (int g = 0; g < 4; ++g) {
      const __bf16* xg = xpcur + g * Hsz + n0 + lo;
#pragma unroll
      for (int r = 0; r < 8; ++r)
        acc[g][r] = (float)xg[(r + 8 * hi) * G4H];
    }

    // Kick off the async fetch of the next step's tile into the other buffer.
#if HAVE_TDM
    if (wave == 0 && t + 1 < Tsz)
      tdm_load_xp(xproj + ((long)(t + 1) * Bsz + b0) * G4H,
                  (unsigned)(LDS_XP0 + pnxt * XP_BYTES));
#else
    if (t + 1 < Tsz) {
      const __bf16* src = xproj + ((long)(t + 1) * Bsz + b0) * G4H;
      __bf16* dst = xp + pnxt * XP_ELEMS;
      for (int i = threadIdx.x * 8; i < XP_ELEMS; i += 1024 * 8)
        *(vu4*)(dst + i) = *(const vu4*)(src + i);
    }
#endif

    // z += h(t-1) @ U : K=512 -> 16 WMMA steps, 4 gate tiles each.
#pragma unroll 2
    for (int kb = 0; kb < Hsz / 32; ++kb) {
      const __bf16* ar = cur + lo * HS + kb * 32 + hi * 16;
      v16bf a = ld2(ar, ar + 8);
#pragma unroll
      for (int g = 0; g < 4; ++g) {
        const __bf16* br = up[g] + kb * 32;
        v16bf bm = ld2(br, br + 8);
        acc[g] = __builtin_amdgcn_wmma_f32_16x16x32_bf16(false, a, false, bm,
                                                         (short)0, acc[g],
                                                         false, false);
      }
    }

    // Gate nonlinearities + cell/hidden update, all in-register.
#pragma unroll
    for (int r = 0; r < 8; ++r) {
      float ig = sigmoidf(acc[0][r]);
      float fg = sigmoidf(acc[1][r]);
      float gg = fmaxf(acc[2][r], 0.0f);     // activation = relu
      float og = sigmoidf(acc[3][r]);
      float cc = fg * c[r] + ig * gg;
      c[r] = cc;
      float h = og * fmaxf(cc, 0.0f);
      int m = r + 8 * hi;
      nxt[m * HS + n0 + lo] = (__bf16)h;
      if (t == Tsz - 1)
        hbf[(long)(b0 + m) * Hsz + n0 + lo] = (__bf16)h;
    }

#if HAVE_TDM
    if (wave == 0) __builtin_amdgcn_s_wait_tensorcnt(0);
#endif
    __syncthreads();
  }
}

// ---------------------------------------------------------------------------
// y = h_last @ Wd + bd : [128,512] x [512,64]. 32 tiles, 4 blocks x 8 waves.
// ---------------------------------------------------------------------------
__global__ __launch_bounds__(256)
void k_dense(const __bf16* __restrict__ hbf,  // [B][H] bf16
             const __bf16* __restrict__ Wdt,  // [FUT][H] bf16 (transposed Wd)
             const float*  __restrict__ bd,   // [FUT]
             float*        __restrict__ out)  // [B][FUT][1]
{
  const int wave = threadIdx.x >> 5;
  const int lane = threadIdx.x & 31;
  const int lo = lane & 15, hi = lane >> 4;
  int tile = blockIdx.x * 8 + wave;           // 0..31
  int mt = tile >> 2, nt = tile & 3;

  const __bf16* arow = hbf + (long)(mt * 16 + lo) * Hsz;
  const __bf16* bcol = Wdt + (long)(nt * 16 + lo) * Hsz;

  v8f acc;
  float bv = bd[nt * 16 + lo];
#pragma unroll
  for (int r = 0; r < 8; ++r) acc[r] = bv;

#pragma unroll 4
  for (int kb = 0; kb < Hsz / 32; ++kb) {
    v16bf a = ld2(arow + kb * 32 + hi * 16, arow + kb * 32 + hi * 16 + 8);
    v16bf w = ld2(bcol + kb * 32 + hi * 16, bcol + kb * 32 + hi * 16 + 8);
    acc = __builtin_amdgcn_wmma_f32_16x16x32_bf16(false, a, false, w,
                                                  (short)0, acc, false, false);
  }

  float* orow = out + (long)(mt * 16) * FUT + nt * 16;
#pragma unroll
  for (int r = 0; r < 8; ++r)
    orow[(r + 8 * hi) * FUT + lo] = acc[r];
}

// ---------------------------------------------------------------------------
extern "C" void kernel_launch(void* const* d_in, const int* in_sizes, int n_in,
                              void* d_out, int out_size, void* d_ws, size_t ws_size,
                              hipStream_t stream) {
  (void)in_sizes; (void)n_in; (void)out_size; (void)ws_size;
  const float* x  = (const float*)d_in[0];   // [B,T,F]
  const float* W  = (const float*)d_in[1];   // [F,4H]
  const float* U  = (const float*)d_in[2];   // [H,4H]
  const float* bb = (const float*)d_in[3];   // [4H]
  const float* Wd = (const float*)d_in[4];   // [H,FUT]
  const float* bd = (const float*)d_in[5];   // [FUT]
  float* out = (float*)d_out;

  char* ws = (char*)d_ws;
  size_t off = 0;
  auto alloc = [&](size_t bytes) -> void* {
    void* p = ws + off;
    off = (off + bytes + 255) & ~(size_t)255;
    return p;
  };

  __bf16* xproj = (__bf16*)alloc((size_t)Tsz * Bsz * G4H * 2);  // 256 MB
  __bf16* xbf   = (__bf16*)alloc((size_t)Bsz * Tsz * Fsz * 2);  // 8 MB
  __bf16* Utb   = (__bf16*)alloc((size_t)G4H * Hsz * 2);        // 2 MB
  __bf16* Wtb   = (__bf16*)alloc((size_t)G4H * Fsz * 2);
  __bf16* Wdtb  = (__bf16*)alloc((size_t)FUT * Hsz * 2);
  __bf16* hbf   = (__bf16*)alloc((size_t)Bsz * Hsz * 2);

  // 1) bf16 conversions (+ transposes so B-operand reads are contiguous)
  {
    int n = Bsz * Tsz * Fsz;
    k_cvt<<<(n + 255) / 256, 256, 0, stream>>>(x, xbf, n);
  }
  k_cvt_t<<<(Fsz * G4H + 255) / 256, 256, 0, stream>>>(W,  Wtb,  Fsz, G4H);
  k_cvt_t<<<(Hsz * G4H + 255) / 256, 256, 0, stream>>>(U,  Utb,  Hsz, G4H);
  k_cvt_t<<<(Hsz * FUT + 255) / 256, 256, 0, stream>>>(Wd, Wdtb, Hsz, FUT);

  // 2) input projection GEMM (17 GFLOP, fills the chip)
  {
    long tiles = (long)(Bsz * Tsz / 16) * (G4H / 16);   // 524288
    k_xproj<<<(int)(tiles / 8), 256, 0, stream>>>(xbf, Wtb, bb, xproj);
  }

  // 3) persistent recurrent scan: 8 blocks x 1024 threads; LDS = xp dbuf + h dbuf
  k_scan<<<Bsz / 16, 1024, LDS_TOTAL, stream>>>(xproj, Utb, hbf);

  // 4) final dense -> d_out
  k_dense<<<4, 256, 0, stream>>>(hbf, Wdtb, bd, out);
}
